// MsfgatModel_75780402970919
// MI455X (gfx1250) — matrix operand order repaired and checked by reading, not compiled, not run on hardware
//
#include <hip/hip_runtime.h>
#include <hip/hip_bf16.h>

#define BATCH 256
#define NN    128
#define FF    133
#define FPAD  160
#define HH    8
#define NHD   64
#define HID   300
#define HIDP  320
#define CAT   512
#define ALPHA 0.2f
#define NEGV  -9.0e15f

typedef __attribute__((ext_vector_type(16))) _Float16 v16h;
typedef __attribute__((ext_vector_type(8)))  float    v8f;

// ---------------- unguarded WMMA fragment loaders (CDNA5 16x16x32 f16 layouts) ----------------
// A 16x32 f16 (row-major source): lane<16 -> row=m0+lane,   K = k0 + {0..7, 16..23}
//                                 lane>=16 -> row=m0+lane-16, K = k0+8 + {0..7, 16..23}
__device__ __forceinline__ v16h a_frag16(const _Float16* __restrict__ A, int ld,
                                         int m0, int k0) {
  int lane = threadIdx.x & 31;
  const _Float16* p = A + (size_t)(m0 + (lane & 15)) * ld + k0 + ((lane >> 4) << 3);
  v16h a;
#pragma unroll
  for (int e = 0; e < 8; ++e) a[e] = p[e];
#pragma unroll
  for (int e = 8; e < 16; ++e) a[e] = p[e + 8];
  return a;
}

// B 32x16 f16 from TRANSPOSED storage Bt[ncols][K] (ldk = K stride):
// lane<16 -> col=n0+lane, K = k0 + {0..7, 16..23} ; lane>=16 -> col=n0+lane-16, K = k0+8 + {0..7, 16..23}
// => two runs of 8 contiguous halves per lane (b128 loads).
__device__ __forceinline__ v16h b_fragT(const _Float16* __restrict__ Bt, int ldk,
                                        int k0, int n0) {
  int lane = threadIdx.x & 31;
  const _Float16* p = Bt + (size_t)(n0 + (lane & 15)) * ldk + k0 + ((lane >> 4) << 3);
  v16h b;
#pragma unroll
  for (int e = 0; e < 16; ++e) b[e] = p[e + ((e >> 3) << 3)];
  return b;
}

#define WMMA_F16(a, b, c) \
  __builtin_amdgcn_wmma_f32_16x16x32_f16(false, (a), false, (b), (short)0, (c), false, false)

// ---------------- padding/transpose kernels (one-time) ----------------
__global__ void k_pad_x(const float* __restrict__ x, _Float16* __restrict__ xp) {
  int idx = blockIdx.x * blockDim.x + threadIdx.x;
  if (idx >= BATCH * NN * FPAD) return;
  int k = idx % FPAD, r = idx / FPAD;
  xp[idx] = (k < FF) ? (_Float16)x[(size_t)r * FF + k] : (_Float16)0.f;
}

// W_heads [H][133][64] -> transposed padded [H][64][160]
__global__ void k_pad_wheads(const float* __restrict__ src, _Float16* __restrict__ dst) {
  int idx = blockIdx.x * blockDim.x + threadIdx.x;
  if (idx >= HH * NHD * FPAD) return;
  int k = idx % FPAD, o = (idx / FPAD) % NHD, h = idx / (FPAD * NHD);
  dst[idx] = (k < FF) ? (_Float16)src[((size_t)h * FF + k) * NHD + o] : (_Float16)0.f;
}

// W_out [512][300] -> transposed padded [320][512]
__global__ void k_pad_wout(const float* __restrict__ src, _Float16* __restrict__ dst) {
  int idx = blockIdx.x * blockDim.x + threadIdx.x;
  if (idx >= HIDP * CAT) return;
  int r = idx % CAT, c = idx / CAT;
  dst[idx] = (c < HID) ? (_Float16)src[(size_t)r * HID + c] : (_Float16)0.f;
}

// ---------------- K1: Wh_t[b,h] (64x128) = (xpad[b] @ Whp_t[h]^T)^T ----------------
// One wave computes a 16x64 output slab; stored transposed for K4's B-fragments.
__global__ void k_gemm_wh(const _Float16* __restrict__ xp, const _Float16* __restrict__ Wp,
                          _Float16* __restrict__ WhT) {
  int wave = blockIdx.x * 4 + (threadIdx.x >> 5);
  int mt = wave & 7, h = (wave >> 3) & 7, b = wave >> 6;
  if (b >= BATCH) return;
  const _Float16* A  = xp + (size_t)b * NN * FPAD;
  const _Float16* Bt = Wp + (size_t)h * NHD * FPAD;
  v8f c0 = {}, c1 = {}, c2 = {}, c3 = {};
  for (int k0 = 0; k0 < FPAD; k0 += 32) {
    v16h a = a_frag16(A, FPAD, mt * 16, k0);
    c0 = WMMA_F16(a, b_fragT(Bt, FPAD, k0,  0), c0);
    c1 = WMMA_F16(a, b_fragT(Bt, FPAD, k0, 16), c1);
    c2 = WMMA_F16(a, b_fragT(Bt, FPAD, k0, 32), c2);
    c3 = WMMA_F16(a, b_fragT(Bt, FPAD, k0, 48), c3);
  }
  int lane  = threadIdx.x & 31;
  int col   = lane & 15;
  int rbase = mt * 16 + ((lane >> 4) << 3);
  _Float16* Od = WhT + (size_t)(b * HH + h) * NHD * NN;
#pragma unroll
  for (int r = 0; r < 8; ++r) {
    int rr = rbase + r;
    Od[(size_t)(col +  0) * NN + rr] = (_Float16)c0[r];
    Od[(size_t)(col + 16) * NN + rr] = (_Float16)c1[r];
    Od[(size_t)(col + 32) * NN + rr] = (_Float16)c2[r];
    Od[(size_t)(col + 48) * NN + rr] = (_Float16)c3[r];
  }
}

// ---------------- K2: GATv2 score vectors src/dst over leaky_relu(Wh) ----------------
// WhT is [bh][o][n]; thread per (bh, n) -> reads are cross-lane coalesced.
__global__ void k_srcdst(const _Float16* __restrict__ WhT, const float* __restrict__ bheads,
                         float* __restrict__ src, float* __restrict__ dst) {
  int idx = blockIdx.x * blockDim.x + threadIdx.x;
  if (idx >= BATCH * HH * NN) return;
  int n = idx % NN, bh = idx / NN, h = bh % HH;
  const _Float16* w = WhT + (size_t)bh * NHD * NN + n;
  const float* bhp = bheads + h * 2 * NHD;
  float s = 0.f, d = 0.f;
#pragma unroll 8
  for (int o = 0; o < NHD; ++o) {
    float v  = (float)w[(size_t)o * NN];
    float lv = v > 0.f ? v : ALPHA * v;
    s += lv * bhp[o];
    d += lv * bhp[NHD + o];
  }
  src[idx] = s; dst[idx] = d;
}

// ---------------- K3: masked-softmax row stats (max, sum-exp), one wave32 per row ----------------
__global__ void k_rowstats(const float* __restrict__ src, const float* __restrict__ dst,
                           const int* __restrict__ adj, float* __restrict__ rmax,
                           float* __restrict__ rsum, int Hd) {
  int wave = blockIdx.x * (blockDim.x >> 5) + (threadIdx.x >> 5);
  int lane = threadIdx.x & 31;
  if (wave >= BATCH * Hd * NN) return;
  int i = wave % NN;
  int b = wave / (Hd * NN);
  int base = (wave / NN) * NN;
  const int* arow = adj + ((size_t)b * NN + i) * NN;
  float si = src[base + i];
  float ev[4], m = -3.4e38f;
#pragma unroll
  for (int q = 0; q < 4; ++q) {
    int j = lane + q * 32;
    ev[q] = (arow[j] > 0) ? (si + dst[base + j]) : NEGV;
    m = fmaxf(m, ev[q]);
  }
  for (int off = 16; off; off >>= 1) m = fmaxf(m, __shfl_xor(m, off, 32));
  float s = 0.f;
#pragma unroll
  for (int q = 0; q < 4; ++q) s += __expf(ev[q] - m);
  for (int off = 16; off; off >>= 1) s += __shfl_xor(s, off, 32);
  if (lane == 0) { rmax[wave] = m; rsum[wave] = s; }
}

// Softmax A-fragment for rows [m0,m0+16) x K [k0,k0+32), built on the fly (unguarded loads).
__device__ __forceinline__ v16h att_a_frag(float si, float mi, float ri,
                                           const int* __restrict__ arow,
                                           const float* __restrict__ dstv,
                                           int k0) {
  int lane = threadIdx.x & 31;
  int kb = k0 + ((lane >> 4) << 3);
  v16h a;
#pragma unroll
  for (int e = 0; e < 16; ++e) {
    int k = kb + (e & 7) + ((e >> 3) << 4);
    float evv = (arow[k] > 0) ? (si + dstv[k]) : NEGV;
    a[e] = (_Float16)(__expf(evv - mi) * ri);
  }
  return a;
}

// ---------------- K4: h = elu(softmax(e) @ Wh); one wave = 16x64 (full head width) ----------------
__global__ void k_att_gemm1(const float* __restrict__ src, const float* __restrict__ dst,
                            const float* __restrict__ rmax, const float* __restrict__ rsum,
                            const int* __restrict__ adj, const _Float16* __restrict__ WhT,
                            _Float16* __restrict__ hcat) {
  int wave = blockIdx.x * 4 + (threadIdx.x >> 5);
  int mt = wave & 7, h = (wave >> 3) & 7, b = wave >> 6;
  if (b >= BATCH) return;
  int lane = threadIdx.x & 31;
  int bh   = b * HH + h;
  int base = bh * NN;
  int row  = mt * 16 + (lane & 15);
  float si = src[base + row];
  float mi = rmax[base + row];
  float ri = 1.0f / rsum[base + row];
  const int* arow = adj + ((size_t)b * NN + row) * NN;
  const float* dstv = dst + base;
  const _Float16* Bt = WhT + (size_t)bh * NHD * NN;
  v8f c0 = {}, c1 = {}, c2 = {}, c3 = {};
  for (int k0 = 0; k0 < NN; k0 += 32) {
    v16h a = att_a_frag(si, mi, ri, arow, dstv, k0);
    c0 = WMMA_F16(a, b_fragT(Bt, NN, k0,  0), c0);
    c1 = WMMA_F16(a, b_fragT(Bt, NN, k0, 16), c1);
    c2 = WMMA_F16(a, b_fragT(Bt, NN, k0, 32), c2);
    c3 = WMMA_F16(a, b_fragT(Bt, NN, k0, 48), c3);
  }
  int col   = lane & 15;
  int rbase = mt * 16 + ((lane >> 4) << 3);
  _Float16* Od = hcat + (size_t)b * NN * CAT + h * NHD;
#pragma unroll
  for (int r = 0; r < 8; ++r) {
    _Float16* p = Od + (size_t)(rbase + r) * CAT + col;
    float v0 = c0[r], v1 = c1[r], v2 = c2[r], v3 = c3[r];
    p[0]  = (_Float16)(v0 > 0.f ? v0 : (__expf(v0) - 1.0f));
    p[16] = (_Float16)(v1 > 0.f ? v1 : (__expf(v1) - 1.0f));
    p[32] = (_Float16)(v2 > 0.f ? v2 : (__expf(v2) - 1.0f));
    p[48] = (_Float16)(v3 > 0.f ? v3 : (__expf(v3) - 1.0f));
  }
}

// ---------------- K5: Wh2_t = (hcat (128x512) @ W_out)^T, padded 320 ----------------
// Padded W_out rows are zero => Wh2 cols 300..319 come out zero automatically.
__global__ void k_gemm_out(const _Float16* __restrict__ hcat, const _Float16* __restrict__ WoT,
                           _Float16* __restrict__ Wh2T) {
  int wave = blockIdx.x * 4 + (threadIdx.x >> 5);
  int ng = wave % 5, mt = (wave / 5) & 7, b = wave / 40;
  if (b >= BATCH) return;
  int n0 = ng * 64;
  const _Float16* A = hcat + (size_t)b * NN * CAT;
  v8f c0 = {}, c1 = {}, c2 = {}, c3 = {};
  for (int k0 = 0; k0 < CAT; k0 += 32) {
    v16h a = a_frag16(A, CAT, mt * 16, k0);
    c0 = WMMA_F16(a, b_fragT(WoT, CAT, k0, n0 +  0), c0);
    c1 = WMMA_F16(a, b_fragT(WoT, CAT, k0, n0 + 16), c1);
    c2 = WMMA_F16(a, b_fragT(WoT, CAT, k0, n0 + 32), c2);
    c3 = WMMA_F16(a, b_fragT(WoT, CAT, k0, n0 + 48), c3);
  }
  int lane  = threadIdx.x & 31;
  int col   = n0 + (lane & 15);
  int rbase = mt * 16 + ((lane >> 4) << 3);
  _Float16* Od = Wh2T + (size_t)b * HIDP * NN;
#pragma unroll
  for (int r = 0; r < 8; ++r) {
    int rr = rbase + r;
    Od[(size_t)(col +  0) * NN + rr] = (_Float16)c0[r];
    Od[(size_t)(col + 16) * NN + rr] = (_Float16)c1[r];
    Od[(size_t)(col + 32) * NN + rr] = (_Float16)c2[r];
    Od[(size_t)(col + 48) * NN + rr] = (_Float16)c3[r];
  }
}

// ---------------- K6: second attention score vectors (Wh2T is [b][feat][node]) ----------------
__global__ void k_srcdst2(const _Float16* __restrict__ Wh2T, const float* __restrict__ bout,
                          float* __restrict__ src2, float* __restrict__ dst2) {
  int idx = blockIdx.x * blockDim.x + threadIdx.x;
  if (idx >= BATCH * NN) return;
  int n = idx % NN, b = idx / NN;
  const _Float16* w = Wh2T + (size_t)b * HIDP * NN + n;
  float s = 0.f, d = 0.f;
  for (int o = 0; o < HID; ++o) {
    float v  = (float)w[(size_t)o * NN];
    float lv = v > 0.f ? v : ALPHA * v;
    s += lv * bout[o];
    d += lv * bout[HID + o];
  }
  src2[idx] = s; dst2[idx] = d;
}

// ---------------- K7: o = elu(softmax(e2) @ Wh2); wave = 16x64 slab ----------------
__global__ void k_att_gemm2(const float* __restrict__ src, const float* __restrict__ dst,
                            const float* __restrict__ rmax, const float* __restrict__ rsum,
                            const int* __restrict__ adj, const _Float16* __restrict__ Wh2T,
                            float* __restrict__ o) {
  int wave = blockIdx.x * 4 + (threadIdx.x >> 5);
  int ng = wave % 5, mt = (wave / 5) & 7, b = wave / 40;
  if (b >= BATCH) return;
  int n0 = ng * 64;
  int lane = threadIdx.x & 31;
  int base = b * NN;
  int row  = mt * 16 + (lane & 15);
  float si = src[base + row];
  float mi = rmax[base + row];
  float ri = 1.0f / rsum[base + row];
  const int* arow = adj + ((size_t)b * NN + row) * NN;
  const float* dstv = dst + base;
  const _Float16* Bt = Wh2T + (size_t)b * HIDP * NN;
  v8f c0 = {}, c1 = {}, c2 = {}, c3 = {};
  for (int k0 = 0; k0 < NN; k0 += 32) {
    v16h a = att_a_frag(si, mi, ri, arow, dstv, k0);
    c0 = WMMA_F16(a, b_fragT(Bt, NN, k0, n0 +  0), c0);
    c1 = WMMA_F16(a, b_fragT(Bt, NN, k0, n0 + 16), c1);
    c2 = WMMA_F16(a, b_fragT(Bt, NN, k0, n0 + 32), c2);
    c3 = WMMA_F16(a, b_fragT(Bt, NN, k0, n0 + 48), c3);
  }
  int col   = n0 + (lane & 15);
  int rbase = mt * 16 + ((lane >> 4) << 3);
  float* Od = o + (size_t)b * NN * HIDP;
#pragma unroll
  for (int r = 0; r < 8; ++r) {
    float* p = Od + (size_t)(rbase + r) * HIDP + col;
    float v0 = c0[r], v1 = c1[r], v2 = c2[r], v3 = c3[r];
    p[0]  = v0 > 0.f ? v0 : (__expf(v0) - 1.0f);
    p[16] = v1 > 0.f ? v1 : (__expf(v1) - 1.0f);
    p[32] = v2 > 0.f ? v2 : (__expf(v2) - 1.0f);
    p[48] = v3 > 0.f ? v3 : (__expf(v3) - 1.0f);
  }
}

// ---------------- K8a: per-(b,i) log-sum-exp over HID features ----------------
__global__ void k_lse(const float* __restrict__ o, float* __restrict__ lse) {
  int wave = blockIdx.x * (blockDim.x >> 5) + (threadIdx.x >> 5);
  int lane = threadIdx.x & 31;
  if (wave >= BATCH * NN) return;
  const float* row = o + (size_t)wave * HIDP;
  float m = -3.4e38f;
  for (int f = lane; f < HID; f += 32) m = fmaxf(m, row[f]);
  for (int off = 16; off; off >>= 1) m = fmaxf(m, __shfl_xor(m, off, 32));
  float s = 0.f;
  for (int f = lane; f < HID; f += 32) s += __expf(row[f] - m);
  for (int off = 16; off; off >>= 1) s += __shfl_xor(s, off, 32);
  if (lane == 0) lse[wave] = m + __logf(s);
}

// ---------------- K8b: g[b,f] = mean_i (o - lse) ----------------
__global__ void k_mean(const float* __restrict__ o, const float* __restrict__ lse,
                       float* __restrict__ g) {
  int idx = blockIdx.x * blockDim.x + threadIdx.x;
  if (idx >= BATCH * HID) return;
  int b = idx / HID, f = idx % HID;
  float acc = 0.f;
  for (int i = 0; i < NN; ++i)
    acc += o[((size_t)b * NN + i) * HIDP + f] - lse[b * NN + i];
  g[idx] = acc * (1.0f / NN);
}

// ---------------- K9: GRU (h0=0) + FFN + sigmoid, one block per molecule ----------------
__global__ void k_head(const float* __restrict__ g, const float* __restrict__ Wi,
                       const float* __restrict__ bi, const float* __restrict__ bh,
                       const float* __restrict__ W1, const float* __restrict__ b1,
                       const float* __restrict__ W2, const float* __restrict__ b2,
                       float* __restrict__ out) {
  __shared__ float sg[HID], sgi[3 * HID], sh[HID], st[HID], sred[128];
  int b = blockIdx.x, tid = threadIdx.x;
  const float* gb = g + (size_t)b * HID;
  for (int j = tid; j < HID; j += blockDim.x) sg[j] = gb[j];
  __syncthreads();
  for (int j = tid; j < 3 * HID; j += blockDim.x) {
    float acc = bi[j];
    const float* wr = Wi + (size_t)j * HID;
    for (int k = 0; k < HID; ++k) acc += sg[k] * wr[k];
    sgi[j] = acc;
  }
  __syncthreads();
  for (int j = tid; j < HID; j += blockDim.x) {
    float r  = 1.f / (1.f + __expf(-(sgi[j] + bh[j])));
    float z  = 1.f / (1.f + __expf(-(sgi[HID + j] + bh[HID + j])));
    float nv = tanhf(sgi[2 * HID + j] + r * bh[2 * HID + j]);
    float hv = (1.f - z) * nv;
    sh[j] = hv > 0.f ? hv : 0.f;
  }
  __syncthreads();
  for (int j = tid; j < HID; j += blockDim.x) {
    float acc = b1[j];
    const float* wr = W1 + (size_t)j * HID;
    for (int k = 0; k < HID; ++k) acc += sh[k] * wr[k];
    st[j] = acc > 0.f ? acc : 0.f;
  }
  __syncthreads();
  float acc = 0.f;
  for (int k = tid; k < HID; k += blockDim.x) acc += st[k] * W2[k];
  sred[tid] = acc;
  __syncthreads();
  for (int w = 64; w; w >>= 1) {
    if (tid < w) sred[tid] += sred[tid + w];
    __syncthreads();
  }
  if (tid == 0) out[b] = 1.f / (1.f + __expf(-(sred[0] + b2[0])));
}

// ---------------- launch ----------------
extern "C" void kernel_launch(void* const* d_in, const int* in_sizes, int n_in,
                              void* d_out, int out_size, void* d_ws, size_t ws_size,
                              hipStream_t stream) {
  const float* x      = (const float*)d_in[0];
  const int*   adj    = (const int*)d_in[1];
  const float* Wheads = (const float*)d_in[2];
  const float* bheads = (const float*)d_in[3];
  const float* Wout   = (const float*)d_in[4];
  const float* bout   = (const float*)d_in[5];
  const float* gWi    = (const float*)d_in[6];
  // d_in[7] = gru_Wh: unused (h0 == 0)
  const float* gbi    = (const float*)d_in[8];
  const float* gbh    = (const float*)d_in[9];
  const float* fW1    = (const float*)d_in[10];
  const float* fb1    = (const float*)d_in[11];
  const float* fW2    = (const float*)d_in[12];
  const float* fb2    = (const float*)d_in[13];
  float* out = (float*)d_out;

  char* w = (char*)d_ws;
  size_t off = 0;
  auto alloc = [&](size_t bytes) -> void* {
    void* p = w + off;
    off = (off + bytes + 255) & ~(size_t)255;
    return p;
  };
  _Float16* xpad  = (_Float16*)alloc((size_t)BATCH * NN * FPAD * sizeof(_Float16));
  _Float16* WhpT  = (_Float16*)alloc((size_t)HH * NHD * FPAD * sizeof(_Float16));
  _Float16* WoutT = (_Float16*)alloc((size_t)HIDP * CAT * sizeof(_Float16));
  _Float16* WhT   = (_Float16*)alloc((size_t)BATCH * HH * NHD * NN * sizeof(_Float16));
  _Float16* hcat  = (_Float16*)alloc((size_t)BATCH * NN * CAT * sizeof(_Float16));
  _Float16* Wh2T  = (_Float16*)alloc((size_t)BATCH * HIDP * NN * sizeof(_Float16));
  float* obuf  = (float*)alloc((size_t)BATCH * NN * HIDP * sizeof(float));
  float* src1  = (float*)alloc((size_t)BATCH * HH * NN * sizeof(float));
  float* dst1  = (float*)alloc((size_t)BATCH * HH * NN * sizeof(float));
  float* rmax1 = (float*)alloc((size_t)BATCH * HH * NN * sizeof(float));
  float* rsum1 = (float*)alloc((size_t)BATCH * HH * NN * sizeof(float));
  float* src2  = (float*)alloc((size_t)BATCH * NN * sizeof(float));
  float* dst2  = (float*)alloc((size_t)BATCH * NN * sizeof(float));
  float* rmax2 = (float*)alloc((size_t)BATCH * NN * sizeof(float));
  float* rsum2 = (float*)alloc((size_t)BATCH * NN * sizeof(float));
  float* lse   = (float*)alloc((size_t)BATCH * NN * sizeof(float));
  float* gbuf  = (float*)alloc((size_t)BATCH * HID * sizeof(float));

  // P: zero-padded (and B-transposed) f16 copies so every WMMA operand load is
  // unconditionally in-bounds AND fully contiguous per lane.
  k_pad_x<<<(BATCH * NN * FPAD + 255) / 256, 256, 0, stream>>>(x, xpad);
  k_pad_wheads<<<(HH * NHD * FPAD + 255) / 256, 256, 0, stream>>>(Wheads, WhpT);
  k_pad_wout<<<(HIDP * CAT + 255) / 256, 256, 0, stream>>>(Wout, WoutT);

  // K1: per-head projection; wave = 16x64 slab (256*8*8 = 16384 waves)
  k_gemm_wh<<<4096, 128, 0, stream>>>(xpad, WhpT, WhT);
  // K2: score vectors
  k_srcdst<<<(BATCH * HH * NN + 255) / 256, 256, 0, stream>>>(WhT, bheads, src1, dst1);
  // K3: softmax row stats (8 waves/block)
  k_rowstats<<<(BATCH * HH * NN + 7) / 8, 256, 0, stream>>>(src1, dst1, adj, rmax1, rsum1, HH);
  // K4: fused masked-softmax @ Wh, elu, head concat; wave = 16x64 slab
  k_att_gemm1<<<4096, 128, 0, stream>>>(src1, dst1, rmax1, rsum1, adj, WhT, hcat);
  // K5: out_att projection; wave = 16x64 slab (256*8*5 = 10240 waves)
  k_gemm_out<<<2560, 128, 0, stream>>>(hcat, WoutT, Wh2T);
  // K6: second attention stats
  k_srcdst2<<<(BATCH * NN + 255) / 256, 256, 0, stream>>>(Wh2T, bout, src2, dst2);
  k_rowstats<<<(BATCH * NN + 7) / 8, 256, 0, stream>>>(src2, dst2, adj, rmax2, rsum2, 1);
  // K7: fused second attention aggregation + elu; wave = 16x64 slab
  k_att_gemm2<<<2560, 128, 0, stream>>>(src2, dst2, rmax2, rsum2, adj, Wh2T, obuf);
  // K8: log-softmax + node mean
  k_lse<<<(BATCH * NN + 7) / 8, 256, 0, stream>>>(obuf, lse);
  k_mean<<<(BATCH * HID + 255) / 256, 256, 0, stream>>>(obuf, lse, gbuf);
  // K9: GRU + FFN + sigmoid
  k_head<<<BATCH, 128, 0, stream>>>(gbuf, gWi, gbi, gbh, fW1, fb1, fW2, fb2, out);

  (void)in_sizes; (void)n_in; (void)out_size; (void)ws_size;
}
